// EntitiesAsExperts_79405355368684
// MI455X (gfx1250) — matrix operand order repaired and checked by reading, not compiled, not run on hardware
//
#include <hip/hip_runtime.h>

// ---------------------------------------------------------------------------
// EntityMemory eval path on MI455X (gfx1250, wave32, WMMA).
// scores = pseudo @ E dominates: 131 GFLOP over a 1.024 GB read of E
// (~44 us at 23.3 TB/s). E converts f32->bf16 after load so the GEMM runs on
// v_wmma_f32_16x16x32_bf16 while HBM traffic stays at the single-read floor.
// Round-2 changes: each wave owns TWO M-tiles (A = 128 VGPRs register
// resident) and a parity subset of N-tiles, so each ds_load_b128 B tile feeds
// two WMMA chains -> LDS B traffic per WMMA halves (the real bottleneck);
// E staging uses float4 (b128) loads.
// Top-k: fused per-128-column-group top-2 candidates + per-mention bitonic
// merge. Workspace: pseudo(256KB) + Apack(128KB) + candidates(~31MB).
// ---------------------------------------------------------------------------

typedef __attribute__((ext_vector_type(16))) __bf16 v16bf;
typedef __attribute__((ext_vector_type(8)))  float  v8f;

#define SEQ   512
#define DE    768
#define DENT  256
#define NENT  1000000
#define NM    256
#define TOPK  100

#define COLS_PER_BLK 8192
#define NBLK2 ((NENT + COLS_PER_BLK - 1) / COLS_PER_BLK)   // 123
#define GROUPS_PER_BLK 64                                   // 8192 / 128
#define NGRP (NBLK2 * GROUPS_PER_BLK)                       // 7872 (>= 7813 real)
#define CAND (NGRP * 2)                                     // 15744 per mention
#define SORTN 16384
#define NEG_INF (-3.4e38f)
#define LDS_STRIDE 132   // 128 + 4 dword pad

__device__ __forceinline__ unsigned f2bf(float f) {
  unsigned u = __float_as_uint(f);
  u += 0x7fffu + ((u >> 16) & 1u);   // round to nearest even
  return u >> 16;
}

// ---------------------------------------------------------------------------
__global__ void k_zero(float* out, int n) {
  int i = blockIdx.x * blockDim.x + threadIdx.x;
  if (i < n) out[i] = 0.0f;
}

// ---------------------------------------------------------------------------
// pseudo[m][d] = bf[d] + sum_i span[m][i] * Wf[d][i]   (span = [first|second])
__global__ void k_pseudo(const float* __restrict__ X,
                         const float* __restrict__ Wf,
                         const float* __restrict__ bf,
                         const int* __restrict__ pos_b,
                         const int* __restrict__ pos_begin,
                         const int* __restrict__ pos_end,
                         float* __restrict__ pseudo) {
  __shared__ float span[2 * DE];
  int m = blockIdx.x, d = threadIdx.x;
  int b = pos_b[m], pb = pos_begin[m], pe = pos_end[m];
  const float* x1 = X + ((size_t)b * SEQ + pb) * DE;
  const float* x2 = X + ((size_t)b * SEQ + pe) * DE;
  for (int i = d; i < DE; i += blockDim.x) { span[i] = x1[i]; span[DE + i] = x2[i]; }
  __syncthreads();
  const float* w = Wf + (size_t)d * (2 * DE);
  float acc = bf[d];
#pragma unroll 4
  for (int i = 0; i < 2 * DE; i++) acc += w[i] * span[i];
  pseudo[m * DENT + d] = acc;
}

// ---------------------------------------------------------------------------
// Repack pseudo into bf16 WMMA A tiles (16x32); lane half h holds k = 16h + s (contiguous).
// Apack[((mt*8+kt)*32 + lane)*8 + j] = packed bf16 pair.
__global__ void k_pack(const float* __restrict__ pseudo,
                       unsigned* __restrict__ Apack) {
  int flat = blockIdx.x * blockDim.x + threadIdx.x;   // 0..32767
  int j  = flat & 7;
  int l  = (flat >> 3) & 31;
  int kt = (flat >> 8) & 7;
  int mt = flat >> 11;
  int m  = mt * 16 + (l & 15);
  int h  = l >> 4;
  int q  = j >> 2;
  int kb = kt * 32 + h * 16 + j * 2;
  unsigned lo = f2bf(pseudo[m * DENT + kb]);
  unsigned hi = f2bf(pseudo[m * DENT + kb + 1]);
  Apack[flat] = lo | (hi << 16);
}

// ---------------------------------------------------------------------------
// Big kernel: scores tile GEMM (bf16 WMMA) + fused top-2-per-128-col-group.
// 512 threads = 16 waves. Wave w owns M-tile PAIR (w&7) and N-tile parity
// (w>>3): each B tile read from LDS is consumed by two WMMAs.
__global__ __launch_bounds__(512) void k_scores(
    const unsigned* __restrict__ Apack,
    const float* __restrict__ E,
    float* __restrict__ candV,
    int* __restrict__ candI) {
  extern __shared__ unsigned lds[];
  unsigned* ldsB = lds;                               // 256 cols * 132 dwords
  float*    S    = (float*)(lds + 256 * LDS_STRIDE);  // 256 m   * 132 floats

  int tid  = threadIdx.x;
  int lane = tid & 31;
  int wv   = tid >> 5;
  int mth  = wv & 7;            // M-tile pair: tiles 2*mth, 2*mth+1
  int ntp  = wv >> 3;           // N-tile parity within a half
  int blk  = blockIdx.x;

  // Two register-resident A tile sets (pseudo, bf16, WMMA layout).
  v16bf a0[8], a1[8];
  const v16bf* ap = (const v16bf*)Apack;
#pragma unroll
  for (int kt = 0; kt < 8; kt++) {
    a0[kt] = ap[((mth * 2)     * 8 + kt) * 32 + lane];
    a1[kt] = ap[((mth * 2 + 1) * 8 + kt) * 32 + lane];
  }

  for (int st = 0; st < 32; st++) {
    int n0 = blk * COLS_PER_BLK + st * 256;
    __syncthreads();
    // Stage E[256 k][256 cols] as packed bf16 pairs; float4 (b128) loads,
    // consecutive threads -> consecutive 16B -> coalesced. NENT%4==0, so a
    // float4 never straddles the boundary.
    for (int it = 0; it < 16; it++) {
      int task = it * 512 + tid;                      // 0..8191
      int cg = task & 63;                             // 4-column group
      int p  = task >> 6;                             // k-pair 0..127
      int col = n0 + cg * 4;
      float4 e0 = make_float4(0.f, 0.f, 0.f, 0.f);
      float4 e1 = make_float4(0.f, 0.f, 0.f, 0.f);
      if (col < NENT) {
        e0 = *(const float4*)&E[(size_t)(2 * p)     * NENT + col];
        e1 = *(const float4*)&E[(size_t)(2 * p + 1) * NENT + col];
      }
      int c0 = cg * 4;
      ldsB[(c0 + 0) * LDS_STRIDE + p] = f2bf(e0.x) | (f2bf(e1.x) << 16);
      ldsB[(c0 + 1) * LDS_STRIDE + p] = f2bf(e0.y) | (f2bf(e1.y) << 16);
      ldsB[(c0 + 2) * LDS_STRIDE + p] = f2bf(e0.z) | (f2bf(e1.z) << 16);
      ldsB[(c0 + 3) * LDS_STRIDE + p] = f2bf(e0.w) | (f2bf(e1.w) << 16);
    }
    __syncthreads();

    for (int half = 0; half < 2; half++) {
      int n_lane = lane & 15, kh = lane >> 4;
#pragma unroll
      for (int ni = 0; ni < 4; ni++) {
        int ntl = ntp + ni * 2;                       // local col tile in half
        v8f accA = {}; v8f accB = {};
#pragma unroll
        for (int kt = 0; kt < 8; kt++) {
          int kbase = kt * 16 + kh * 8;               // dword idx, 16B aligned
          const v16bf* bp = (const v16bf*)&ldsB[(size_t)((half * 8 + ntl) * 16 + n_lane) * LDS_STRIDE + kbase];
          v16bf b = *bp;
          accA = __builtin_amdgcn_wmma_f32_16x16x32_bf16(false, a0[kt], false, b,
                                                         (short)0, accA, false, false);
          accB = __builtin_amdgcn_wmma_f32_16x16x32_bf16(false, a1[kt], false, b,
                                                         (short)0, accB, false, false);
        }
        int rowA = (mth * 2) * 16 + kh * 8;           // C layout: VGPR r -> M row
        int rowB = rowA + 16;
#pragma unroll
        for (int r = 0; r < 8; r++) {
          S[(rowA + r) * LDS_STRIDE + ntl * 16 + n_lane] = accA[r];
          S[(rowB + r) * LDS_STRIDE + ntl * 16 + n_lane] = accB[r];
        }
      }
      __syncthreads();
      // top-2 per (mention, 128-col group)
      if (tid < NM) {
        int m = tid;
        int base = n0 + half * 128;
        float v1 = NEG_INF, v2 = NEG_INF; int i1 = 0, i2 = 0;
        for (int i = 0; i < 128; i++) {
          int col = base + i;
          if (col < NENT) {
            float v = S[m * LDS_STRIDE + i];
            if (v > v1)      { v2 = v1; i2 = i1; v1 = v; i1 = col; }
            else if (v > v2) { v2 = v;  i2 = col; }
          }
        }
        int gg = blk * GROUPS_PER_BLK + st * 2 + half;
        size_t o = (size_t)m * CAND + (size_t)gg * 2;
        candV[o]     = v1; candI[o]     = i1;
        candV[o + 1] = v2; candI[o + 1] = i2;
      }
      __syncthreads();
    }
  }
}

// ---------------------------------------------------------------------------
// Per-mention: bitonic-sort candidates (desc), softmax top-100, gather E
// columns, project through Wb, scatter to y[pos_b, pos_begin].
__global__ __launch_bounds__(512) void k_merge(
    const float* __restrict__ candV, const int* __restrict__ candI,
    const float* __restrict__ E, const float* __restrict__ Wb,
    const float* __restrict__ bb,
    const int* __restrict__ pos_b, const int* __restrict__ pos_begin,
    float* __restrict__ out) {
  extern __shared__ unsigned lds3[];
  float* sv     = (float*)lds3;          // SORTN
  int*   si     = (int*)(sv + SORTN);    // SORTN
  float* alpha  = (float*)(si + SORTN);  // 128
  float* picked = alpha + 128;           // 256
  int m = blockIdx.x, tid = threadIdx.x;

  for (int i = tid; i < SORTN; i += 512) {
    if (i < CAND) { sv[i] = candV[(size_t)m * CAND + i]; si[i] = candI[(size_t)m * CAND + i]; }
    else          { sv[i] = NEG_INF; si[i] = 0; }
  }
  __syncthreads();

  for (unsigned k = 2; k <= SORTN; k <<= 1) {
    for (unsigned j = k >> 1; j > 0; j >>= 1) {
      for (unsigned i = tid; i < SORTN; i += 512) {
        unsigned ixj = i ^ j;
        if (ixj > i) {
          bool desc = ((i & k) == 0);
          float av = sv[i], bv = sv[ixj];
          if (desc ? (av < bv) : (av > bv)) {
            int ai = si[i];
            sv[i] = bv; sv[ixj] = av;
            si[i] = si[ixj]; si[ixj] = ai;
          }
        }
      }
      __syncthreads();
    }
  }

  if (tid == 0) {  // softmax over top-100 (tiny, serial)
    float mx = sv[0];
    for (int i = 1; i < TOPK; i++) mx = fmaxf(mx, sv[i]);
    float s = 0.f;
    for (int i = 0; i < TOPK; i++) { float e = __expf(sv[i] - mx); alpha[i] = e; s += e; }
    float inv = 1.f / s;
    for (int i = 0; i < TOPK; i++) alpha[i] *= inv;
  }
  __syncthreads();

  if (tid < DENT) {  // picked[d] = sum_j alpha_j * E[d][idx_j]
    float acc = 0.f;
    for (int j = 0; j < TOPK; j++) acc += alpha[j] * E[(size_t)tid * NENT + si[j]];
    picked[tid] = acc;
  }
  __syncthreads();

  int b = pos_b[m], pb = pos_begin[m];
  float* yrow = out + ((size_t)b * SEQ + pb) * DE;
  for (int e = tid; e < DE; e += 512) {
    const float* w = Wb + (size_t)e * DENT;
    float acc = bb[e];
#pragma unroll 4
    for (int d = 0; d < DENT; d++) acc += w[d] * picked[d];
    yrow[e] = acc;
  }
}

// ---------------------------------------------------------------------------
extern "C" void kernel_launch(void* const* d_in, const int* in_sizes, int n_in,
                              void* d_out, int out_size, void* d_ws, size_t ws_size,
                              hipStream_t stream) {
  (void)in_sizes; (void)n_in; (void)ws_size;
  const float* X         = (const float*)d_in[0];
  const float* Wf        = (const float*)d_in[1];
  const float* bf        = (const float*)d_in[2];
  const float* Wb        = (const float*)d_in[3];
  const float* bb        = (const float*)d_in[4];
  const float* E         = (const float*)d_in[5];
  const int*   pos_b     = (const int*)d_in[6];
  const int*   pos_begin = (const int*)d_in[7];
  const int*   pos_end   = (const int*)d_in[8];
  // d_in[9] is k (=100), compile-time TOPK here.
  float* out = (float*)d_out;

  char* ws = (char*)d_ws;
  float*    pseudo = (float*)ws;                                   // 256 KB
  unsigned* Apack  = (unsigned*)(ws + 262144);                     // 128 KB
  float*    candV  = (float*)(ws + 262144 + 131072);
  int*      candI  = (int*)((char*)candV + sizeof(float) * (size_t)NM * CAND);

  k_zero<<<(out_size + 1023) / 1024, 1024, 0, stream>>>(out, out_size);
  k_pseudo<<<NM, 256, 0, stream>>>(X, Wf, bf, pos_b, pos_begin, pos_end, pseudo);
  k_pack<<<128, 256, 0, stream>>>(pseudo, Apack);

  size_t lds2 = (size_t)(256 * LDS_STRIDE) * 4 * 2;                // 264.2 KB
  k_scores<<<NBLK2, 512, lds2, stream>>>(Apack, E, candV, candI);

  size_t lds3 = (size_t)SORTN * 8 + (128 + 256) * 4;               // ~129.5 KB
  k_merge<<<NM, 512, lds3, stream>>>(candV, candI, E, Wb, bb, pos_b, pos_begin, out);
}
